// TransformerBlock_6322191860209
// MI455X (gfx1250) — matrix operand order repaired and checked by reading, not compiled, not run on hardware
//
#include <hip/hip_runtime.h>
#include <hip/hip_bf16.h>
#include <stdint.h>

// Problem constants (match reference)
#define Bq 2
#define Nq 8192
#define KK 16
#define CH 64

typedef __attribute__((ext_vector_type(16))) __bf16        v16bf;
typedef __attribute__((ext_vector_type(8)))  float         v8f;
typedef __attribute__((ext_vector_type(8)))  unsigned int  v8u;

// Async global->LDS path (guarded: falls back to uint4 copies if absent).
// The gfx1250 builtin takes generic pointers to "int __vector(4)" (vector_size
// GCC-style), per the compiler's own diagnostic.
#if defined(__gfx1250__) && __has_builtin(__builtin_amdgcn_global_load_async_to_lds_b128) && \
    __has_builtin(__builtin_amdgcn_s_wait_asynccnt)
#define HAVE_ASYNC_LDS 1
typedef int async_v4i __attribute__((vector_size(16)));
#define ASYNC_PTR(p) ((async_v4i*)(p))
#else
#define HAVE_ASYNC_LDS 0
#endif

// ---------- bf16 helpers (native converts; ushort raw storage) ----------
__device__ __forceinline__ unsigned short f2bf_bits(float f) {
    return __builtin_bit_cast(unsigned short, (__bf16)f);
}
__device__ __forceinline__ float bf2f(unsigned short h) {
    return __builtin_bit_cast(float, (unsigned)h << 16);
}

// ---------- WMMA wrapper ----------
__device__ __forceinline__ v8f wmma_bf16(v16bf a, v16bf b, v8f c) {
    return __builtin_amdgcn_wmma_f32_16x16x32_bf16(
        /*neg_a=*/false, a, /*neg_b=*/false, b,
        /*c_mod=*/(short)0, c, /*reuse_a=*/false, /*reuse_b=*/false);
}

// ---------- Swizzled fragment storage ----------
// Fragments are kept in LDS as [frag][lane][16 bf16] so a fragment load is one
// contiguous 32-byte read per lane (2x ds_load_b128).
__device__ __forceinline__ v16bf load_frag_swz(const unsigned short* buf, int f) {
    unsigned lane = threadIdx.x & 31u;
    return __builtin_bit_cast(v16bf, *(const v8u*)(buf + f * 512 + lane * 16));
}

// Store a 16x16 f32 D-fragment (cols n0..n0+15) into an A-swizzled bf16 buffer
// covering a 16x64 matrix (2 fragment pages of 512 ushorts).
// A-frag element e of consumer lane (m, half_a) maps to k = 16*(e>=8) + 8*half_a + (e&7).
__device__ __forceinline__ void store_d_swz(unsigned short* buf, int n0, v8f c, bool relu) {
    unsigned lane = threadIdx.x & 31u;
    int n = lane & 15, half_d = lane >> 4;
    int k = n0 + n;
    int fa = k >> 5, kk = k & 31;
    int e = ((kk >> 4) << 3) + (kk & 7);
    int half_a = (kk >> 3) & 1;
    unsigned short* p = buf + fa * 512 + half_a * 256 + (8 * half_d) * 16 + e;
#pragma unroll
    for (int r = 0; r < 8; ++r) {
        float v = c[r];
        if (relu) v = v > 0.f ? v : 0.f;
        p[r * 16] = f2bf_bits(v);
    }
}

// Build a 16x32 bf16 A-fragment; functor f(m, kb) returns float2 for (kb, kb+1).
template <typename F>
__device__ __forceinline__ v16bf make_a_frag(int k0, F f) {
    unsigned lane = threadIdx.x & 31u;
    int m = lane & 15, half = lane >> 4;
    v16bf a;
#pragma unroll
    for (int v = 0; v < 8; ++v) {
        int kb = k0 + (v >> 2) * 16 + half * 8 + (v & 3) * 2;
        float2 t = f(m, kb);
        a[2 * v]     = (__bf16)t.x;
        a[2 * v + 1] = (__bf16)t.y;
    }
    return a;
}

// Swizzle-convert a 64x64 f32 weight matrix (row-major, k-major) into B-fragment
// order: flat index i -> frag f=i>>9 (f = nt*2+kt), lane=(i>>4)&31, e=i&15;
// source element (k = kt*32 + (e>>3)*16 + (lane>>4)*8 + (e&7), n = nt*16 + (lane&15)).
__device__ __forceinline__ int wswz_src(int i) {
    int f = i >> 9, lane = (i >> 4) & 31, e = i & 15;
    int nt = f >> 1, kt = f & 1;
    int n = nt * 16 + (lane & 15);
    int k = kt * 32 + (e >> 3) * 16 + (lane >> 4) * 8 + (e & 7);
    return k * CH + n;
}

// ============================================================================
// Kernel 1: brute-force KNN (top-16 smallest squared distances, self included)
// ============================================================================
#define CT 512
__global__ __launch_bounds__(128) void knn_kernel(const float* __restrict__ pos,
                                                  int* __restrict__ knn) {
    __shared__ float tile[CT * 3];
    const int bb = blockIdx.y;
    const int q  = blockIdx.x * 128 + threadIdx.x;
    const float* pb = pos + (size_t)bb * Nq * 3;
    const float px = pb[q * 3 + 0], py = pb[q * 3 + 1], pz = pb[q * 3 + 2];

    float dist[KK];
    int   idx[KK];
#pragma unroll
    for (int i = 0; i < KK; ++i) { dist[i] = 3.4e38f; idx[i] = 0; }

    for (int t = 0; t < Nq / CT; ++t) {
        __syncthreads();
        for (int i = threadIdx.x; i < CT * 3; i += 128) tile[i] = pb[t * CT * 3 + i];
        __syncthreads();
        if (t + 1 < Nq / CT) __builtin_prefetch(&pb[(t + 1) * CT * 3 + threadIdx.x * 12], 0, 0);
        for (int j = 0; j < CT; ++j) {
            float dx = px - tile[j * 3 + 0];
            float dy = py - tile[j * 3 + 1];
            float dz = pz - tile[j * 3 + 2];
            float d = dx * dx + dy * dy + dz * dz;
            if (d < dist[KK - 1]) {
                dist[KK - 1] = d; idx[KK - 1] = t * CT + j;
#pragma unroll
                for (int s = KK - 1; s > 0; --s) {
                    if (dist[s] < dist[s - 1]) {
                        float td = dist[s]; dist[s] = dist[s - 1]; dist[s - 1] = td;
                        int   ti = idx[s];  idx[s]  = idx[s - 1];  idx[s - 1]  = ti;
                    }
                }
            }
        }
    }
    int* o = knn + ((size_t)bb * Nq + q) * KK;
#pragma unroll
    for (int i = 0; i < KK; ++i) o[i] = idx[i];
}

// ============================================================================
// Kernel 2: x = features@emb_w+emb_b; q = x@wq (f32); k = x@wk, v = x@wv (bf16)
// ============================================================================
__global__ __launch_bounds__(256) void embed_qkv_kernel(
    const float* __restrict__ features,
    const float* __restrict__ emb_wg, const float* __restrict__ emb_bg,
    const float* __restrict__ wq_g, const float* __restrict__ wk_g,
    const float* __restrict__ wv_g,
    float* __restrict__ qg, unsigned short* __restrict__ kg,
    unsigned short* __restrict__ vg) {
    __shared__ alignas(32) unsigned short w_emb[CH * CH];  // B-frag swizzled
    __shared__ alignas(32) unsigned short w_q[CH * CH];
    __shared__ alignas(32) unsigned short w_k[CH * CH];
    __shared__ alignas(32) unsigned short w_v[CH * CH];
    __shared__ alignas(16) float b_emb[CH];
    __shared__ alignas(32) unsigned short s_x[8][16 * CH];  // A-frag swizzled bounce

    const int tid = threadIdx.x;
    for (int i = tid; i < CH * CH; i += blockDim.x) {
        int s = wswz_src(i);
        w_emb[i] = f2bf_bits(emb_wg[s]);
        w_q[i]   = f2bf_bits(wq_g[s]);
        w_k[i]   = f2bf_bits(wk_g[s]);
        w_v[i]   = f2bf_bits(wv_g[s]);
    }
    for (int i = tid; i < CH; i += blockDim.x) b_emb[i] = emb_bg[i];
    __syncthreads();

    const int w = tid >> 5, lane = tid & 31, m = lane & 15, half = lane >> 4;
    const size_t r0 = (size_t)(blockIdx.x * 8 + w) * 16;

    auto ffun = [&](int mm, int kb) { return *(const float2*)&features[(r0 + mm) * CH + kb]; };
    v16bf f0 = make_a_frag(0, ffun);
    v16bf f1 = make_a_frag(32, ffun);

    // x = features @ emb_w + emb_b  -> A-swizzled bf16 bounce tile
#pragma unroll
    for (int nt = 0; nt < 4; ++nt) {
        v8f c; float bias = b_emb[nt * 16 + m];
#pragma unroll
        for (int r = 0; r < 8; ++r) c[r] = bias;
        c = wmma_bf16(f0, load_frag_swz(w_emb, nt * 2 + 0), c);
        c = wmma_bf16(f1, load_frag_swz(w_emb, nt * 2 + 1), c);
        store_d_swz(s_x[w], nt * 16, c, false);
    }
    __syncthreads();

    v16bf x0 = load_frag_swz(s_x[w], 0);
    v16bf x1 = load_frag_swz(s_x[w], 1);

    // q (f32 out)
#pragma unroll
    for (int nt = 0; nt < 4; ++nt) {
        int n0 = nt * 16;
        v8f c;
#pragma unroll
        for (int r = 0; r < 8; ++r) c[r] = 0.f;
        c = wmma_bf16(x0, load_frag_swz(w_q, nt * 2 + 0), c);
        c = wmma_bf16(x1, load_frag_swz(w_q, nt * 2 + 1), c);
#pragma unroll
        for (int r = 0; r < 8; ++r) qg[(r0 + r + 8 * half) * CH + n0 + m] = c[r];
    }
    // k (bf16 out)
#pragma unroll
    for (int nt = 0; nt < 4; ++nt) {
        int n0 = nt * 16;
        v8f c;
#pragma unroll
        for (int r = 0; r < 8; ++r) c[r] = 0.f;
        c = wmma_bf16(x0, load_frag_swz(w_k, nt * 2 + 0), c);
        c = wmma_bf16(x1, load_frag_swz(w_k, nt * 2 + 1), c);
#pragma unroll
        for (int r = 0; r < 8; ++r) kg[(r0 + r + 8 * half) * CH + n0 + m] = f2bf_bits(c[r]);
    }
    // v (bf16 out)
#pragma unroll
    for (int nt = 0; nt < 4; ++nt) {
        int n0 = nt * 16;
        v8f c;
#pragma unroll
        for (int r = 0; r < 8; ++r) c[r] = 0.f;
        c = wmma_bf16(x0, load_frag_swz(w_v, nt * 2 + 0), c);
        c = wmma_bf16(x1, load_frag_swz(w_v, nt * 2 + 1), c);
#pragma unroll
        for (int r = 0; r < 8; ++r) vg[(r0 + r + 8 * half) * CH + n0 + m] = f2bf_bits(c[r]);
    }
}

// ============================================================================
// Kernel 3: per-query attention. One wave per query (16 neighbors = M=16 tile).
// ============================================================================
#define AW 4  // waves (= queries) per block
__global__ __launch_bounds__(128) void pt_attn_kernel(
    const float* __restrict__ pos, const float* __restrict__ features,
    const float* __restrict__ pe_w1g, const float* __restrict__ pe_b1g,
    const float* __restrict__ pe_w2g, const float* __restrict__ pe_b2g,
    const float* __restrict__ at_w1g, const float* __restrict__ at_b1g,
    const float* __restrict__ at_w2g, const float* __restrict__ at_b2g,
    const float* __restrict__ out_wg, const float* __restrict__ out_bg,
    const int* __restrict__ knn, const float* __restrict__ qg,
    const unsigned short* __restrict__ kg, const unsigned short* __restrict__ vg,
    float* __restrict__ out) {
    __shared__ alignas(32) unsigned short w_pe2[CH * CH];   // B-frag swizzled
    __shared__ alignas(32) unsigned short w_at1[CH * CH];   // B-frag swizzled
    __shared__ alignas(32) unsigned short w_at2[CH * CH];   // B-frag swizzled
    __shared__ alignas(32) unsigned short w_outT[CH * CH];  // transposed [c][k]
    __shared__ alignas(16) float w_pe1[3 * CH];
    __shared__ alignas(16) float b_pe1[CH], b_pe2[CH], b_at1[CH], b_at2[CH], b_out[CH];
    __shared__ alignas(32) unsigned short s_k[AW][KK * CH];   // gathered k; reused as swizzled hidden
    __shared__ alignas(32) unsigned short s_v[AW][KK * CH];   // gathered v (row-major)
    __shared__ alignas(32) unsigned short s_pe[AW][KK * CH];  // posenc (row-major)
    __shared__ alignas(16) float s_q[AW][CH];
    __shared__ float s_g[AW][KK * 3];
    __shared__ int   s_idx[AW][KK];
    __shared__ float s_res[AW][CH];

    const int tid = threadIdx.x;
    for (int i = tid; i < CH * CH; i += blockDim.x) {
        int s = wswz_src(i);
        w_pe2[i] = f2bf_bits(pe_w2g[s]);
        w_at1[i] = f2bf_bits(at_w1g[s]);
        w_at2[i] = f2bf_bits(at_w2g[s]);
        int c = i >> 6, k = i & 63;  // transpose out_w for contiguous per-channel reads
        w_outT[i] = f2bf_bits(out_wg[k * CH + c]);
    }
    for (int i = tid; i < 3 * CH; i += blockDim.x) w_pe1[i] = pe_w1g[i];
    for (int i = tid; i < CH; i += blockDim.x) {
        b_pe1[i] = pe_b1g[i]; b_pe2[i] = pe_b2g[i];
        b_at1[i] = at_b1g[i]; b_at2[i] = at_b2g[i]; b_out[i] = out_bg[i];
    }

    const int w = tid >> 5, lane = tid & 31, m = lane & 15, half = lane >> 4;
    const int gq = blockIdx.x * AW + w;
    const int bb = gq / Nq;

    if (lane < KK) s_idx[w][lane] = knn[(size_t)gq * KK + lane];
    s_q[w][lane]      = qg[(size_t)gq * CH + lane];
    s_q[w][lane + 32] = qg[(size_t)gq * CH + lane + 32];
    const float px = pos[(size_t)gq * 3 + 0];
    const float py = pos[(size_t)gq * 3 + 1];
    const float pz = pos[(size_t)gq * 3 + 2];
    __syncthreads();

    // --- gather grouped_pos, k, v for this query's 16 neighbors ---
    {
        int j = s_idx[w][m];
        if (lane < KK) {
            s_g[w][m * 3 + 0] = pos[((size_t)bb * Nq + j) * 3 + 0] - px;
            s_g[w][m * 3 + 1] = pos[((size_t)bb * Nq + j) * 3 + 1] - py;
            s_g[w][m * 3 + 2] = pos[((size_t)bb * Nq + j) * 3 + 2] - pz;
        }
        const int coff = half * 32;  // lanes 0-15: ch 0-31 of row m; lanes 16-31: ch 32-63
        const unsigned short* ksrc = kg + ((size_t)bb * Nq + j) * CH + coff;
        const unsigned short* vsrc = vg + ((size_t)bb * Nq + j) * CH + coff;
        unsigned short* kdst = &s_k[w][m * CH + coff];
        unsigned short* vdst = &s_v[w][m * CH + coff];
#if HAVE_ASYNC_LDS
#pragma unroll
        for (int i = 0; i < 4; ++i) {
            __builtin_amdgcn_global_load_async_to_lds_b128(ASYNC_PTR(ksrc + i * 8), ASYNC_PTR(kdst + i * 8), 0, 0);
            __builtin_amdgcn_global_load_async_to_lds_b128(ASYNC_PTR(vsrc + i * 8), ASYNC_PTR(vdst + i * 8), 0, 0);
        }
        __builtin_amdgcn_s_wait_asynccnt(0);
#else
#pragma unroll
        for (int i = 0; i < 4; ++i) {
            ((uint4*)kdst)[i] = ((const uint4*)ksrc)[i];
            ((uint4*)vdst)[i] = ((const uint4*)vsrc)[i];
        }
#endif
    }
    __syncthreads();

    // --- posenc layer 1 (3->64, ReLU) built directly in A-fragment layout ---
    const float gx = s_g[w][m * 3 + 0], gy = s_g[w][m * 3 + 1], gz = s_g[w][m * 3 + 2];
    auto pe1 = [&](int mm, int kb) {
        (void)mm;
        float2 w0 = *(const float2*)&w_pe1[kb];
        float2 w1 = *(const float2*)&w_pe1[CH + kb];
        float2 w2 = *(const float2*)&w_pe1[2 * CH + kb];
        float2 b2 = *(const float2*)&b_pe1[kb];
        float2 r;
        r.x = fmaxf(gx * w0.x + gy * w1.x + gz * w2.x + b2.x, 0.f);
        r.y = fmaxf(gx * w0.y + gy * w1.y + gz * w2.y + b2.y, 0.f);
        return r;
    };
    v16bf p0 = make_a_frag(0, pe1);
    v16bf p1 = make_a_frag(32, pe1);

    // --- posenc layer 2: pe = relu1 @ pe_w2 + b_pe2 -> s_pe (row-major bf16) ---
#pragma unroll
    for (int nt = 0; nt < 4; ++nt) {
        int n0 = nt * 16;
        v8f c; float bias = b_pe2[n0 + m];
#pragma unroll
        for (int r = 0; r < 8; ++r) c[r] = bias;
        c = wmma_bf16(p0, load_frag_swz(w_pe2, nt * 2 + 0), c);
        c = wmma_bf16(p1, load_frag_swz(w_pe2, nt * 2 + 1), c);
#pragma unroll
        for (int r = 0; r < 8; ++r) s_pe[w][(r + 8 * half) * CH + n0 + m] = f2bf_bits(c[r]);
    }
    __syncthreads();

    // --- h = q - k + posenc -> A fragments (pairwise b32/b64 loads) ---
    auto hfun = [&](int mm, int kb) {
        float2 q2 = *(const float2*)&s_q[w][kb];
        unsigned kv = *(const unsigned*)&s_k[w][mm * CH + kb];
        unsigned pv = *(const unsigned*)&s_pe[w][mm * CH + kb];
        float2 r;
        r.x = q2.x - bf2f((unsigned short)kv) + bf2f((unsigned short)pv);
        r.y = q2.y - bf2f((unsigned short)(kv >> 16)) + bf2f((unsigned short)(pv >> 16));
        return r;
    };
    v16bf h0 = make_a_frag(0, hfun);
    v16bf h1 = make_a_frag(32, hfun);
    __syncthreads();  // all lanes done reading s_k before it is reused

    // --- attn layer 1: relu(h @ at_w1 + b_at1) -> s_k reused as swizzled hidden ---
#pragma unroll
    for (int nt = 0; nt < 4; ++nt) {
        v8f c; float bias = b_at1[nt * 16 + m];
#pragma unroll
        for (int r = 0; r < 8; ++r) c[r] = bias;
        c = wmma_bf16(h0, load_frag_swz(w_at1, nt * 2 + 0), c);
        c = wmma_bf16(h1, load_frag_swz(w_at1, nt * 2 + 1), c);
        store_d_swz(s_k[w], nt * 16, c, true);
    }
    __syncthreads();

    v16bf a0 = load_frag_swz(s_k[w], 0);
    v16bf a1 = load_frag_swz(s_k[w], 1);

    // --- attn layer 2 + softmax over neighbors + weighted sum ---
    // D-frag: lane holds column n = m, rows r + 8*half; the other 8 rows of the
    // same column live in lane^16 -> reduce with __shfl_xor(.,16).
#pragma unroll
    for (int nt = 0; nt < 4; ++nt) {
        int n0 = nt * 16;
        v8f c; float bias = b_at2[n0 + m];
#pragma unroll
        for (int r = 0; r < 8; ++r) c[r] = bias;
        c = wmma_bf16(a0, load_frag_swz(w_at2, nt * 2 + 0), c);
        c = wmma_bf16(a1, load_frag_swz(w_at2, nt * 2 + 1), c);

        float mx = -3.4e38f;
#pragma unroll
        for (int r = 0; r < 8; ++r) { c[r] *= 0.125f; mx = fmaxf(mx, c[r]); }
        mx = fmaxf(mx, __shfl_xor(mx, 16, 32));
        float e[8], sum = 0.f;
#pragma unroll
        for (int r = 0; r < 8; ++r) { e[r] = __expf(c[r] - mx); sum += e[r]; }
        sum += __shfl_xor(sum, 16, 32);
        const float inv = 1.f / sum;
        float acc = 0.f;
#pragma unroll
        for (int r = 0; r < 8; ++r) {
            int row = r + 8 * half;
            float wv = bf2f(s_v[w][row * CH + n0 + m]) + bf2f(s_pe[w][row * CH + n0 + m]);
            acc += e[r] * inv * wv;
        }
        acc += __shfl_xor(acc, 16, 32);
        if (half == 0) s_res[w][n0 + m] = acc;
    }
    __syncthreads();

    // --- out projection + residual (w_outT gives contiguous per-channel rows) ---
#pragma unroll
    for (int t = 0; t < 2; ++t) {
        int cidx = lane + t * 32;
        float acc = b_out[cidx] + features[(size_t)gq * CH + cidx];
#pragma unroll 8
        for (int k = 0; k < CH; ++k) acc += s_res[w][k] * bf2f(w_outT[cidx * CH + k]);
        out[(size_t)gq * CH + cidx] = acc;
    }
}

// ============================================================================
extern "C" void kernel_launch(void* const* d_in, const int* in_sizes, int n_in,
                              void* d_out, int out_size, void* d_ws, size_t ws_size,
                              hipStream_t stream) {
    (void)in_sizes; (void)n_in; (void)out_size; (void)ws_size;
    const float* pos      = (const float*)d_in[0];
    const float* features = (const float*)d_in[1];
    const float* emb_w    = (const float*)d_in[2];
    const float* emb_b    = (const float*)d_in[3];
    const float* wq       = (const float*)d_in[4];
    const float* wk       = (const float*)d_in[5];
    const float* wv       = (const float*)d_in[6];
    const float* pe_w1    = (const float*)d_in[7];
    const float* pe_b1    = (const float*)d_in[8];
    const float* pe_w2    = (const float*)d_in[9];
    const float* pe_b2    = (const float*)d_in[10];
    const float* at_w1    = (const float*)d_in[11];
    const float* at_b1    = (const float*)d_in[12];
    const float* at_w2    = (const float*)d_in[13];
    const float* at_b2    = (const float*)d_in[14];
    const float* out_w    = (const float*)d_in[15];
    const float* out_b    = (const float*)d_in[16];

    // Workspace layout (total 9 MB):
    //   [0, 1MB)  knn idx  int32 [B*N*16]
    //   [1, 5MB)  q        f32   [B*N*64]
    //   [5, 7MB)  k        bf16  [B*N*64]
    //   [7, 9MB)  v        bf16  [B*N*64]
    char* ws = (char*)d_ws;
    int*            knn = (int*)ws;
    float*          qg  = (float*)(ws + ((size_t)1 << 20));
    unsigned short* kg  = (unsigned short*)(ws + ((size_t)5 << 20));
    unsigned short* vg  = (unsigned short*)(ws + ((size_t)7 << 20));

    knn_kernel<<<dim3(Nq / 128, Bq), 128, 0, stream>>>(pos, knn);
    embed_qkv_kernel<<<dim3((Bq * Nq) / 128), 256, 0, stream>>>(
        features, emb_w, emb_b, wq, wk, wv, qg, kg, vg);
    pt_attn_kernel<<<dim3((Bq * Nq) / AW), 128, 0, stream>>>(
        pos, features, pe_w1, pe_b1, pe_w2, pe_b2,
        at_w1, at_b1, at_w2, at_b2, out_w, out_b,
        knn, qg, kg, vg, (float*)d_out);
}
